// TransformerDecoderLayer_56444460204216
// MI455X (gfx1250) — matrix operand order repaired and checked by reading, not compiled, not run on hardware
//
#include <hip/hip_runtime.h>
#include <math.h>

// ---------------- problem constants ----------------
#define E_DIM   1024
#define NHEADS  16
#define HD      64
#define TSEQ    1024
#define BDIM    4
#define ROWS    (TSEQ * BDIM)      // 4096 flattened (L,B) rows
#define FF_DIM  4096

// ---------------- vector types (ext_vector: trivially constructible) ----
typedef __attribute__((ext_vector_type(16))) _Float16     v16h;
typedef __attribute__((ext_vector_type(8)))  float        v8f;
typedef __attribute__((ext_vector_type(4)))  _Float16     h4;
typedef __attribute__((ext_vector_type(4)))  float        f32x4;
typedef __attribute__((ext_vector_type(4)))  unsigned int u32x4;
typedef __attribute__((ext_vector_type(4)))  int          i32x4;

union Frag { u32x4 q[2]; v16h v; };

// ---------------- async global->LDS copy (ASYNCcnt path), guarded -------
#if defined(__HIP_DEVICE_COMPILE__) && defined(__has_builtin)
#  if __has_builtin(__builtin_amdgcn_global_load_async_to_lds_b128)
#    define HAVE_ASYNC_LDS 1
#  endif
#endif
#ifndef HAVE_ASYNC_LDS
#  define HAVE_ASYNC_LDS 0
#endif

#if HAVE_ASYNC_LDS
typedef __attribute__((address_space(1))) i32x4 gvec_t;   // global int4
typedef __attribute__((address_space(3))) i32x4 lvec_t;   // LDS int4
#endif

__device__ __forceinline__ void cp16_g2l(const _Float16* gp, _Float16* lp)
{
#if HAVE_ASYNC_LDS
    // flat LDS pointer: low 32 bits are the LDS offset (ISA aperture rule)
    __builtin_amdgcn_global_load_async_to_lds_b128(
        (gvec_t*)(unsigned long long)(size_t)gp,
        (lvec_t*)(unsigned int)(size_t)lp,
        0, 0);
#else
    *(u32x4*)lp = *(const u32x4*)gp;
#endif
}

__device__ __forceinline__ void wait_async()
{
#if HAVE_ASYNC_LDS
#  if __has_builtin(__builtin_amdgcn_s_wait_asynccnt)
    __builtin_amdgcn_s_wait_asynccnt(0);
#  else
    asm volatile("s_wait_asynccnt 0x0" ::: "memory");
#  endif
#endif
}

// =====================================================================
// WMMA GEMM on f16 operands, f32 accumulation.
//   BT=true : opB = B^T,  B stored [N,K] row-major  (projections/scores/FFN)
//   BT=false: opB = B,    B stored [K,N] row-major  (attn @ V)
// Outputs: C (f32, optional) and/or C16 (f16, optional), same layout/ldc.
// Batched via blockIdx.z: z=(b<<4)|h, base offset = b*Out + h*In.
// Block = 128 threads (4 waves); tile 64x64; K step 64 (2 x 32-deep subs).
// Each wave: 32x32 patch -> 8 x v_wmma_f32_16x16x32_f16 per K step.
// =====================================================================
template <bool BT>
__global__ __launch_bounds__(128) void wmma_gemm(
    const _Float16* __restrict__ A, const _Float16* __restrict__ B,
    float* __restrict__ C, _Float16* __restrict__ C16,
    const float* __restrict__ bias, const float* __restrict__ residual,
    int M, int N, int K, int lda, int ldb, int ldc,
    long long aOut, long long aIn, long long bOut, long long bIn,
    long long cOut, long long cIn, float scale, int relu)
{
    __shared__ __align__(16) _Float16 As[64][64];   // [m][k]
    __shared__ __align__(16) _Float16 Bs[64][64];   // [n][k]

    const int z  = blockIdx.z;
    const int zb = z >> 4, zh = z & 15;
    A += (size_t)zb * aOut + (size_t)zh * aIn;
    B += (size_t)zb * bOut + (size_t)zh * bIn;
    const size_t coff = (size_t)zb * cOut + (size_t)zh * cIn;
    if (C)   C   += coff;
    if (C16) C16 += coff;

    const int m0 = blockIdx.x * 64;
    const int n0 = blockIdx.y * 64;

    const int lane = threadIdx.x & 31;
    const int wave = threadIdx.x >> 5;
    const int wm   = (wave & 1) * 32;
    const int wn   = (wave >> 1) * 32;
    const int r    = lane & 15;
    const int hi   = lane >> 4;

    v8f acc[2][2] = {};

    for (int k0 = 0; k0 < K; k0 += 64) {
        if (k0 + 64 < K) {   // near-cache prefetch of next K tile
            __builtin_prefetch(A + (size_t)(m0 + (threadIdx.x >> 1)) * lda + k0 + 64, 0, 3);
            if (BT)
                __builtin_prefetch(B + (size_t)(n0 + (threadIdx.x >> 1)) * ldb + k0 + 64, 0, 3);
        }

        // ---- fill As: 64x64 f16, 512 x 16B copies over 128 threads ----
        #pragma unroll
        for (int it = 0; it < 4; ++it) {
            int idx = (int)threadIdx.x + it * 128;
            int row = idx >> 3;
            int c8  = (idx & 7) << 3;
            cp16_g2l(A + (size_t)(m0 + row) * lda + k0 + c8, &As[row][c8]);
        }
        // ---- fill Bs as [n][k] ----
        if (BT) {
            #pragma unroll
            for (int it = 0; it < 4; ++it) {
                int idx = (int)threadIdx.x + it * 128;
                int row = idx >> 3;
                int c8  = (idx & 7) << 3;
                cp16_g2l(B + (size_t)(n0 + row) * ldb + k0 + c8, &Bs[row][c8]);
            }
        } else {
            #pragma unroll
            for (int it = 0; it < 4; ++it) {
                int idx = (int)threadIdx.x + it * 128;
                int kr = idx >> 3;            // 0..63
                int n8 = (idx & 7) << 3;      // 0..56
                u32x4 raw = *(const u32x4*)(B + (size_t)(k0 + kr) * ldb + n0 + n8);
                const _Float16* hh = (const _Float16*)&raw;
                #pragma unroll
                for (int t = 0; t < 8; ++t) Bs[n8 + t][kr] = hh[t];
            }
        }
        wait_async();
        __syncthreads();

        // ---- two 32-deep WMMA sub-steps per LDS fill ----
        #pragma unroll
        for (int s = 0; s < 2; ++s) {
            const int kb = s * 32;
            v16h af[2], bf[2];
            #pragma unroll
            for (int i = 0; i < 2; ++i) {
                Frag fa;
                // lane<16: K kb+0..7 / kb+16..23 ; lane>=16: kb+8..15 / kb+24..31
                fa.q[0] = *(const u32x4*)&As[wm + i * 16 + r][kb + hi * 8];
                fa.q[1] = *(const u32x4*)&As[wm + i * 16 + r][kb + 16 + hi * 8];
                af[i] = fa.v;
            }
            #pragma unroll
            for (int j = 0; j < 2; ++j) {
                Frag fb;
                const u32x4* pb = (const u32x4*)&Bs[wn + j * 16 + r][kb + hi * 8];
                fb.q[0] = pb[0];
                fb.q[1] = pb[2];
                bf[j] = fb.v;
            }
            #pragma unroll
            for (int i = 0; i < 2; ++i)
                #pragma unroll
                for (int j = 0; j < 2; ++j)
                    acc[i][j] = __builtin_amdgcn_wmma_f32_16x16x32_f16(
                        false, af[i], false, bf[j], (short)0, acc[i][j], false, false);
        }
        __syncthreads();
    }

    // ---- epilogue: scale, bias, residual, relu, dual-precision store ----
    #pragma unroll
    for (int i = 0; i < 2; ++i) {
        #pragma unroll
        for (int j = 0; j < 2; ++j) {
            const int gcol = n0 + wn + j * 16 + r;
            #pragma unroll
            for (int rr = 0; rr < 8; ++rr) {
                const int grow = m0 + wm + i * 16 + hi * 8 + rr;
                float v = acc[i][j][rr] * scale;
                if (bias)     v += bias[gcol];
                if (residual) v += residual[(size_t)grow * ldc + gcol];
                if (relu)     v = fmaxf(v, 0.0f);
                if (C)   C[(size_t)grow * ldc + gcol]   = v;
                if (C16) C16[(size_t)grow * ldc + gcol] = (_Float16)v;
            }
        }
    }
}

// ============ fp32 -> f16 conversion (vector x4) ======================
__global__ void cvt_f16_kernel(const float* __restrict__ src,
                               _Float16* __restrict__ dst, int n4)
{
    int i = blockIdx.x * blockDim.x + threadIdx.x;
    if (i >= n4) return;
    f32x4 f = *(const f32x4*)(src + 4 * (size_t)i);
    h4 h = { (_Float16)f.x, (_Float16)f.y, (_Float16)f.z, (_Float16)f.w };
    *(h4*)(dst + 4 * (size_t)i) = h;
}

// =====================================================================
// pos_embed (f16 out; feeds GEMM A operands only):
//   out[..,2t]   = cos(p*(2t+1)*w)*x[..,2t+1]
//   out[..,2t+1] = sin(p*(2t)*w)  *x[..,2t],   w = pi/E
// =====================================================================
__global__ void pos_embed_kernel(const float* __restrict__ x,
                                 const float* __restrict__ pos,
                                 _Float16* __restrict__ out, int npairs)
{
    int idx = blockIdx.x * blockDim.x + threadIdx.x;
    if (idx >= npairs) return;
    const int halfE = E_DIM >> 1;
    int rb = idx / halfE;         // l*B + b
    int t  = idx - rb * halfE;
    float p     = pos[rb];
    float omega = 3.14159265358979323846f / (float)E_DIM;
    const float* xr = x + (size_t)rb * E_DIM;
    _Float16* outr  = out + (size_t)rb * E_DIM;
    outr[2 * t]     = (_Float16)(cosf(p * (float)(2 * t + 1) * omega) * xr[2 * t + 1]);
    outr[2 * t + 1] = (_Float16)(sinf(p * (float)(2 * t)     * omega) * xr[2 * t]);
}

// ===== row softmax: one wave/row; writes fp32 in place + f16 copy =====
__global__ void softmax_kernel(float* __restrict__ x, _Float16* __restrict__ x16,
                               int rows, int cols)
{
    int row  = blockIdx.x * (blockDim.x >> 5) + (threadIdx.x >> 5);
    int lane = threadIdx.x & 31;
    if (row >= rows) return;
    float* p      = x   + (size_t)row * cols;
    _Float16* p16 = x16 + (size_t)row * cols;
    float m = -3.4e38f;
    for (int c = lane; c < cols; c += 32) m = fmaxf(m, p[c]);
    for (int o = 16; o; o >>= 1) m = fmaxf(m, __shfl_xor(m, o, 32));
    float s = 0.0f;
    for (int c = lane; c < cols; c += 32) { float v = expf(p[c] - m); p[c] = v; s += v; }
    for (int o = 16; o; o >>= 1) s += __shfl_xor(s, o, 32);
    float inv = 1.0f / s;
    for (int c = lane; c < cols; c += 32) {
        float v = p[c] * inv;
        p[c]   = v;
        p16[c] = (_Float16)v;
    }
}

// == layernorm (unbiased, ddof=1) in place, optional f16 dual-write ====
__global__ void layernorm_kernel(float* __restrict__ x, _Float16* __restrict__ x16,
                                 const float* __restrict__ g,
                                 const float* __restrict__ b, int rows, int cols)
{
    int row  = blockIdx.x * (blockDim.x >> 5) + (threadIdx.x >> 5);
    int lane = threadIdx.x & 31;
    if (row >= rows) return;
    float* p      = x + (size_t)row * cols;
    _Float16* p16 = x16 ? x16 + (size_t)row * cols : (_Float16*)0;
    float s = 0.0f, ss = 0.0f;
    for (int c = lane; c < cols; c += 32) { float v = p[c]; s += v; ss += v * v; }
    for (int o = 16; o; o >>= 1) { s += __shfl_xor(s, o, 32); ss += __shfl_xor(ss, o, 32); }
    float n    = (float)cols;
    float mean = s / n;
    float var  = (ss - n * mean * mean) / (n - 1.0f);
    float inv  = rsqrtf(var + 1e-5f);
    for (int c = lane; c < cols; c += 32) {
        float o = g[c] * (p[c] - mean) * inv + b[c];
        p[c] = o;
        if (p16) p16[c] = (_Float16)o;
    }
}

// ======== ws = attn.mean(heads): [b,h,l,s] -> [b,l,s] (fp32) ==========
__global__ void mean_heads_kernel(const float* __restrict__ attn,
                                  float* __restrict__ out, int n)
{
    int idx = blockIdx.x * blockDim.x + threadIdx.x;
    if (idx >= n) return;
    int b  = idx >> 20;                 // L*S = 2^20
    int ls = idx & 0xFFFFF;
    const float* base = attn + (size_t)(b * NHEADS) * 1048576ull + ls;
    float s = 0.0f;
    #pragma unroll
    for (int h = 0; h < NHEADS; ++h) s += base[(size_t)h * 1048576ull];
    out[idx] = s * (1.0f / (float)NHEADS);
}

// =====================================================================
extern "C" void kernel_launch(void* const* d_in, const int* in_sizes, int n_in,
                              void* d_out, int out_size, void* d_ws, size_t ws_size,
                              hipStream_t stream)
{
    const float* tgt    = (const float*)d_in[0];
    const float* memory = (const float*)d_in[1];
    const float* pos    = (const float*)d_in[2];
    const float* qpos   = (const float*)d_in[3];
    const float* wqkv_s = (const float*)d_in[4];
    const float* bqkv_s = (const float*)d_in[5];
    const float* wo_s   = (const float*)d_in[6];
    const float* bo_s   = (const float*)d_in[7];
    const float* wqkv_c = (const float*)d_in[8];
    const float* bqkv_c = (const float*)d_in[9];
    const float* wo_c   = (const float*)d_in[10];
    const float* bo_c   = (const float*)d_in[11];
    const float* w1     = (const float*)d_in[12];
    const float* b1     = (const float*)d_in[13];
    const float* w2     = (const float*)d_in[14];
    const float* b2     = (const float*)d_in[15];
    const float* g1     = (const float*)d_in[16];
    const float* be1    = (const float*)d_in[17];
    const float* g2     = (const float*)d_in[18];
    const float* be2    = (const float*)d_in[19];

    float* out_tgt = (float*)d_out;                       // (T,B,E) = 4M f32
    float* out_ws  = out_tgt + (size_t)ROWS * E_DIM;      // (B,T,S) = 4M f32

    const size_t MAT = (size_t)ROWS * E_DIM;              // 4,194,304
    const long long EE = (long long)E_DIM * E_DIM;
    const long long LS = (long long)TSEQ * TSEQ;          // 1,048,576

    // ---- workspace carve-up: f32 first, then f16 (all sizes 16B mult) ----
    char* wsb = (char*)d_ws;
    auto allocF = [&](size_t n) { float* p = (float*)wsb; wsb += n * sizeof(float); return p; };
    auto allocH = [&](size_t n) { _Float16* p = (_Float16*)wsb; wsb += n * sizeof(_Float16); return p; };

    float* S  = allocF((size_t)BDIM * NHEADS * TSEQ * TSEQ);   // 64M f32
    float* X1 = allocF(MAT);
    float* X2 = allocF(MAT);

    _Float16* qk16     = allocH(MAT);
    _Float16* mk16     = allocH(MAT);
    _Float16* tgt16    = allocH(MAT);
    _Float16* mem16    = allocH(MAT);
    _Float16* Q16      = allocH(MAT);
    _Float16* K16      = allocH(MAT);
    _Float16* V16      = allocH(MAT);
    _Float16* S16      = allocH((size_t)BDIM * NHEADS * TSEQ * TSEQ);
    _Float16* ctx16    = allocH(MAT);
    _Float16* X2_16    = allocH(MAT);
    _Float16* H16      = allocH((size_t)ROWS * FF_DIM);
    _Float16* wqkv_s16 = allocH(3 * (size_t)EE);
    _Float16* wo_s16   = allocH((size_t)EE);
    _Float16* wqkv_c16 = allocH(3 * (size_t)EE);
    _Float16* wo_c16   = allocH((size_t)EE);
    _Float16* w1_16    = allocH((size_t)FF_DIM * E_DIM);
    _Float16* w2_16    = allocH((size_t)FF_DIM * E_DIM);

    auto cvt = [&](const float* s, _Float16* d, size_t n) {
        int n4 = (int)(n / 4);
        cvt_f16_kernel<<<n4 / 256, 256, 0, stream>>>(s, d, n4);
    };
    auto gemmNT = [&](const _Float16* A, const _Float16* B, float* C, _Float16* C16,
                      const float* bias, const float* res, int M, int N, int K,
                      int lda, int ldb, int ldc,
                      long long aO, long long aI, long long bO, long long bI,
                      long long cO, long long cI, float scale, int relu, int batch) {
        dim3 g(M / 64, N / 64, batch);
        wmma_gemm<true><<<g, 128, 0, stream>>>(A, B, C, C16, bias, res, M, N, K,
                                               lda, ldb, ldc, aO, aI, bO, bI, cO, cI,
                                               scale, relu);
    };
    auto gemmNN = [&](const _Float16* A, const _Float16* B, float* C, _Float16* C16,
                      int M, int N, int K, int lda, int ldb, int ldc,
                      long long aO, long long aI, long long bO, long long bI,
                      long long cO, long long cI, int batch) {
        dim3 g(M / 64, N / 64, batch);
        wmma_gemm<false><<<g, 128, 0, stream>>>(A, B, C, C16, nullptr, nullptr, M, N, K,
                                                lda, ldb, ldc, aO, aI, bO, bI, cO, cI,
                                                1.0f, 0);
    };

    const int NP = (int)(MAT / 2);      // pos-embed pairs
    const float ISQ = 0.125f;           // 1/sqrt(64)
    const int SMROWS = BDIM * NHEADS * TSEQ;

    // ---- one-time f16 masters (activations + weights) ----
    cvt(tgt, tgt16, MAT);
    cvt(memory, mem16, MAT);
    cvt(wqkv_s, wqkv_s16, 3 * (size_t)EE);
    cvt(wo_s, wo_s16, (size_t)EE);
    cvt(wqkv_c, wqkv_c16, 3 * (size_t)EE);
    cvt(wo_c, wo_c16, (size_t)EE);
    cvt(w1, w1_16, (size_t)FF_DIM * E_DIM);
    cvt(w2, w2_16, (size_t)FF_DIM * E_DIM);

    // ================= self attention =================
    pos_embed_kernel<<<NP / 256, 256, 0, stream>>>(tgt, qpos, qk16, NP);
    gemmNT(qk16,  wqkv_s16,          nullptr, Q16, bqkv_s,             nullptr,
           ROWS, E_DIM, E_DIM, E_DIM, E_DIM, E_DIM, 0,0, 0,0, 0,0, 1.0f, 0, 1);
    gemmNT(qk16,  wqkv_s16 + EE,     nullptr, K16, bqkv_s + E_DIM,     nullptr,
           ROWS, E_DIM, E_DIM, E_DIM, E_DIM, E_DIM, 0,0, 0,0, 0,0, 1.0f, 0, 1);
    gemmNT(tgt16, wqkv_s16 + 2 * EE, nullptr, V16, bqkv_s + 2 * E_DIM, nullptr,
           ROWS, E_DIM, E_DIM, E_DIM, E_DIM, E_DIM, 0,0, 0,0, 0,0, 1.0f, 0, 1);
    // scores[b,h,l,s] = Qp @ Kp^T * ISQ  (64 batches over (L,B,E) slices)
    gemmNT(Q16, K16, S, nullptr, nullptr, nullptr, TSEQ, TSEQ, HD,
           BDIM * E_DIM, BDIM * E_DIM, TSEQ,
           E_DIM, HD, E_DIM, HD, NHEADS * LS, LS, ISQ, 0, BDIM * NHEADS);
    softmax_kernel<<<SMROWS / 8, 256, 0, stream>>>(S, S16, SMROWS, TSEQ);
    gemmNN(S16, V16, nullptr, ctx16, TSEQ, HD, TSEQ, TSEQ, BDIM * E_DIM, BDIM * E_DIM,
           NHEADS * LS, LS, E_DIM, HD, E_DIM, HD, BDIM * NHEADS);
    gemmNT(ctx16, wo_s16, X1, nullptr, bo_s, tgt,
           ROWS, E_DIM, E_DIM, E_DIM, E_DIM, E_DIM, 0,0, 0,0, 0,0, 1.0f, 0, 1);
    layernorm_kernel<<<ROWS / 8, 256, 0, stream>>>(X1, nullptr, g1, be1, ROWS, E_DIM);

    // ================= cross attention =================
    pos_embed_kernel<<<NP / 256, 256, 0, stream>>>(X1, qpos, qk16, NP);
    pos_embed_kernel<<<NP / 256, 256, 0, stream>>>(memory, pos, mk16, NP);
    gemmNT(qk16,  wqkv_c16,          nullptr, Q16, bqkv_c,             nullptr,
           ROWS, E_DIM, E_DIM, E_DIM, E_DIM, E_DIM, 0,0, 0,0, 0,0, 1.0f, 0, 1);
    gemmNT(mk16,  wqkv_c16 + EE,     nullptr, K16, bqkv_c + E_DIM,     nullptr,
           ROWS, E_DIM, E_DIM, E_DIM, E_DIM, E_DIM, 0,0, 0,0, 0,0, 1.0f, 0, 1);
    gemmNT(mem16, wqkv_c16 + 2 * EE, nullptr, V16, bqkv_c + 2 * E_DIM, nullptr,
           ROWS, E_DIM, E_DIM, E_DIM, E_DIM, E_DIM, 0,0, 0,0, 0,0, 1.0f, 0, 1);
    gemmNT(Q16, K16, S, nullptr, nullptr, nullptr, TSEQ, TSEQ, HD,
           BDIM * E_DIM, BDIM * E_DIM, TSEQ,
           E_DIM, HD, E_DIM, HD, NHEADS * LS, LS, ISQ, 0, BDIM * NHEADS);
    softmax_kernel<<<SMROWS / 8, 256, 0, stream>>>(S, S16, SMROWS, TSEQ);
    mean_heads_kernel<<<(BDIM * TSEQ * TSEQ) / 256, 256, 0, stream>>>(
        S, out_ws, BDIM * TSEQ * TSEQ);
    gemmNN(S16, V16, nullptr, ctx16, TSEQ, HD, TSEQ, TSEQ, BDIM * E_DIM, BDIM * E_DIM,
           NHEADS * LS, LS, E_DIM, HD, E_DIM, HD, BDIM * NHEADS);
    gemmNT(ctx16, wo_c16, X2, nullptr, bo_c, X1,
           ROWS, E_DIM, E_DIM, E_DIM, E_DIM, E_DIM, 0,0, 0,0, 0,0, 1.0f, 0, 1);
    layernorm_kernel<<<ROWS / 8, 256, 0, stream>>>(X2, X2_16, g2, be2, ROWS, E_DIM);

    // ================= FFN =================
    gemmNT(X2_16, w1_16, nullptr, H16, b1, nullptr,
           ROWS, FF_DIM, E_DIM, E_DIM, E_DIM, FF_DIM, 0,0, 0,0, 0,0, 1.0f, 1, 1);
    gemmNT(H16, w2_16, out_tgt, nullptr, b2, X2,
           ROWS, E_DIM, FF_DIM, FF_DIM, FF_DIM, E_DIM, 0,0, 0,0, 0,0, 1.0f, 0, 1);
}
